// GAT_61186104099647
// MI455X (gfx1250) — matrix operand order repaired and checked by reading, not compiled, not run on hardware
//
#include <hip/hip_runtime.h>
#include <hip/hip_bf16.h>
#include <cstdint>
#include <math.h>

#define N_NODES 8192
#define F_IN    128
#define F_OUT   64

typedef __bf16 bf16x16 __attribute__((ext_vector_type(16)));
typedef __bf16 bf16x4  __attribute__((ext_vector_type(4)));
typedef float  f32x8   __attribute__((ext_vector_type(8)));
typedef float  f32x4   __attribute__((ext_vector_type(4)));
typedef int    i32x4   __attribute__((ext_vector_type(4)));
// pointer-to-addrspace typedefs for the async-copy builtin (signature discovered
// from the compiler diagnostic: param0 is "int4 __device__ *")
typedef __attribute__((address_space(1))) i32x4* as1_i32x4_ptr;
typedef __attribute__((address_space(3))) i32x4* as3_i32x4_ptr;

// workspace layout (bytes)
#define WS_XT    (0u)                          // bf16 [F_IN][N]   : 2 MB
#define WS_MASK  (2u*1024*1024)                // u32  [N][256]    : 8 MB
#define WS_AGG   (10u*1024*1024)               // f32  [N][F_OUT]  : 2 MB
#define WS_AGGT  (12u*1024*1024)               // bf16 [F_OUT][N]  : 1 MB
#define WS_SJ    (13u*1024*1024)               // f32  [N]
#define WS_E     (13u*1024*1024 + 64u*1024)    // f32  [N]

#if defined(__has_builtin)
#if __has_builtin(__builtin_amdgcn_global_load_async_to_lds_b128)
#define USE_ASYNC_E 1
#endif
#endif

// Split workgroup barrier: only waits for our LDS stores (DScnt), leaving global
// loads/stores in flight across the barrier (unlike __syncthreads' full drain).
__device__ __forceinline__ void wg_split_barrier() {
  asm volatile("s_wait_dscnt 0x0\n\t"
               "s_barrier_signal -1\n\t"
               "s_barrier_wait -1" ::: "memory");
}

// ---------------- K0: X (f32, row-major) -> XT (bf16, col-major) ----------------
__global__ void k0_xt(const float* __restrict__ X, __bf16* __restrict__ XT) {
  int idx = blockIdx.x * 256 + threadIdx.x;     // N*F_IN threads
  int r = idx & (N_NODES - 1);
  int c = idx >> 13;
  XT[(size_t)c * N_NODES + r] = (__bf16)X[(size_t)r * F_IN + c];
}

// ---------------- K1: agg = (A@X)@W + bias ; emit (A+I != 0) bitmask ----------------
// Block: 256 threads (8 waves), 32 output rows. K-loop: 128 cols of A per iter,
// software pipelined: A tile kk+1 and B regs for kk stay in flight across the
// split barrier while WMMA(kk) runs out of the double-buffered LDS tile.
// Mask word convention: word (row, g, e) bit l  <->  col = 128*g + 4*l + e.
__global__ __launch_bounds__(256) void k1_agg(const float* __restrict__ A,
                                              const __bf16* __restrict__ XT,
                                              const float* __restrict__ W,
                                              const float* __restrict__ bias,
                                              uint32_t* __restrict__ mask,
                                              float* __restrict__ agg) {
  __shared__ __bf16 lds_a[2][32 * 136];  // double-buffered A tile 32x128 bf16
  __shared__ float  lds_w[F_IN * F_OUT];
  __shared__ float  lds_ax[32 * 132];    // AX tile f32 for stage 2
  const int t = threadIdx.x;
  const int w = t >> 5;                  // wave id 0..7 -> N-tile of X (16 cols each)
  const int l = t & 31;
  const int row0 = blockIdx.x * 32;

  for (int i = t; i < F_IN * F_OUT; i += 256) lds_w[i] = W[i];

  f32x8 acc0 = {}; f32x8 acc1 = {};
  const int arow = l & 15;               // A-operand: both lane halves map M=0..15
  const int kb   = (l >> 4) * 8;         // K sub-base for A operand halves
  const int bn   = w * 16 + (l & 15);    // B-operand column
  const int bko  = (l >> 4) * 8;         // K sub-base for B operand halves

  // prologue: load A tile 0 (non-temporal: A is streamed exactly once)
  f32x4 areg[2][4];
  #pragma unroll
  for (int it = 0; it < 4; ++it)
    areg[0][it] = __builtin_nontemporal_load(
        (const f32x4*)(A + (size_t)(row0 + w + 8 * it) * N_NODES + 4 * l));

  #pragma unroll 2
  for (int kk = 0; kk < N_NODES / 128; ++kk) {
    const int k = kk * 128;
    __bf16* buf = &lds_a[kk & 1][0];
    // ---- stage tile kk: ballot mask + bf16 convert into LDS ----
    uint32_t bw[4][4];
    #pragma unroll
    for (int it = 0; it < 4; ++it) {
      const int rl = w + 8 * it;
      const int rg = row0 + rl;
      f32x4 a = areg[kk & 1][it];
      uint32_t b0 = __builtin_amdgcn_ballot_w32(a.x != 0.f);
      uint32_t b1 = __builtin_amdgcn_ballot_w32(a.y != 0.f);
      uint32_t b2 = __builtin_amdgcn_ballot_w32(a.z != 0.f);
      uint32_t b3 = __builtin_amdgcn_ballot_w32(a.w != 0.f);
      // diagonal of (A+I): wave-uniform single-bit OR
      const uint32_t uoff = (uint32_t)(rg - k);
      const uint32_t dbit = (uoff < 128u) ? (1u << (uoff >> 2)) : 0u;
      bw[it][0] = b0 | (((uoff & 3u) == 0u) ? dbit : 0u);
      bw[it][1] = b1 | (((uoff & 3u) == 1u) ? dbit : 0u);
      bw[it][2] = b2 | (((uoff & 3u) == 2u) ? dbit : 0u);
      bw[it][3] = b3 | (((uoff & 3u) == 3u) ? dbit : 0u);
      bf16x4 ab = {(__bf16)a.x, (__bf16)a.y, (__bf16)a.z, (__bf16)a.w};
      *(bf16x4*)(buf + rl * 136 + 4 * l) = ab;
    }
    if (l == 0) {  // single guarded region: 4 b128 mask stores
      #pragma unroll
      for (int it = 0; it < 4; ++it) {
        uint4 mw; mw.x = bw[it][0]; mw.y = bw[it][1]; mw.z = bw[it][2]; mw.w = bw[it][3];
        *(uint4*)(mask + (size_t)(row0 + w + 8 * it) * 256 + kk * 4) = mw;
      }
    }
    // ---- issue A loads for tile kk+1 (stay in flight across the barrier) ----
    if (kk + 1 < N_NODES / 128) {
      #pragma unroll
      for (int it = 0; it < 4; ++it)
        areg[(kk + 1) & 1][it] = __builtin_nontemporal_load(
            (const f32x4*)(A + (size_t)(row0 + w + 8 * it) * N_NODES +
                           (kk + 1) * 128 + 4 * l));
    }
    // ---- preload B operands for tile kk (independent of LDS) ----
    bf16x16 bmat[4];
    #pragma unroll
    for (int s = 0; s < 4; ++s) {
      const __bf16* pb = XT + (size_t)bn * N_NODES + k + 32 * s + bko;
      #pragma unroll
      for (int e2 = 0; e2 < 8; ++e2) { bmat[s][e2] = pb[e2]; bmat[s][e2 + 8] = pb[16 + e2]; }
    }
    wg_split_barrier();
    // ---- 4 WMMA K-substeps of 32 ----
    #pragma unroll
    for (int s = 0; s < 4; ++s) {
      const int ks = 32 * s;
      bf16x16 amat0, amat1;
      const __bf16* pa0 = buf + (arow)*136 + ks + kb;
      const __bf16* pa1 = buf + (arow + 16) * 136 + ks + kb;
      #pragma unroll
      for (int e2 = 0; e2 < 8; ++e2) {
        amat0[e2] = pa0[e2]; amat0[e2 + 8] = pa0[16 + e2];
        amat1[e2] = pa1[e2]; amat1[e2 + 8] = pa1[16 + e2];
      }
      acc0 = __builtin_amdgcn_wmma_f32_16x16x32_bf16(false, amat0, false, bmat[s],
                                                     (short)0, acc0, false, false);
      acc1 = __builtin_amdgcn_wmma_f32_16x16x32_bf16(false, amat1, false, bmat[s],
                                                     (short)0, acc1, false, false);
    }
  }

  __syncthreads();
  // spill AX tiles to LDS (C layout: VGPR r -> M=r (lanes 0-15) / M=r+8 (lanes 16-31))
  {
    const int hb = (l >> 4) * 8;
    const int c = w * 16 + (l & 15);
    #pragma unroll
    for (int r = 0; r < 8; ++r) {
      lds_ax[(r + hb) * 132 + c]      = acc0[r];
      lds_ax[(r + hb + 16) * 132 + c] = acc1[r];
    }
  }
  __syncthreads();
  // stage 2: agg = AX @ W + bias, exact f32 (tiny GEMM)
  {
    const int r  = t >> 3;
    const int cg = (t & 7) * 8;
    float s[8] = {0, 0, 0, 0, 0, 0, 0, 0};
    for (int kx = 0; kx < F_IN; ++kx) {
      float av = lds_ax[r * 132 + kx];
      const float* wr = lds_w + kx * F_OUT + cg;
      #pragma unroll
      for (int c = 0; c < 8; ++c) s[c] = fmaf(av, wr[c], s[c]);
    }
    float* og = agg + (size_t)(row0 + r) * F_OUT + cg;
    #pragma unroll
    for (int c = 0; c < 8; ++c) og[c] = s[c] + bias[cg + c];
  }
}

// ---------------- K2a: sj[i] = agg[i,:] . phi_j ----------------
__global__ void k2a_sj(const float* __restrict__ agg, const float* __restrict__ phi,
                       float* __restrict__ sj) {
  int i = blockIdx.x * 256 + threadIdx.x;
  const float* ar = agg + (size_t)i * F_OUT;
  const float* pj = phi + F_OUT;  // phi[f:, 0]
  float s = 0.f;
  #pragma unroll
  for (int c = 0; c < F_OUT; ++c) s = fmaf(ar[c], pj[c], s);
  sj[i] = s;
}

// ---------------- K2b: smax = max(sj); e = exp(sj - smax) ----------------
__global__ __launch_bounds__(256) void k2b_exp(const float* __restrict__ sj,
                                               float* __restrict__ e) {
  __shared__ float red[256];
  int t = threadIdx.x;
  float m = -INFINITY;
  for (int i = t; i < N_NODES; i += 256) m = fmaxf(m, sj[i]);
  red[t] = m;
  __syncthreads();
  for (int off = 128; off; off >>= 1) {
    if (t < off) red[t] = fmaxf(red[t], red[t + off]);
    __syncthreads();
  }
  float smax = red[0];
  for (int i = t; i < N_NODES; i += 256) e[i] = __expf(sj[i] - smax);
}

// ---------------- K2c: agg (f32 row-major) -> aggT (bf16 col-major) ----------------
__global__ void k2c_aggT(const float* __restrict__ agg, __bf16* __restrict__ aggT) {
  int idx = blockIdx.x * 256 + threadIdx.x;     // N*F_OUT threads
  int r = idx & (N_NODES - 1);
  int c = idx >> 13;
  aggT[(size_t)c * N_NODES + r] = (__bf16)agg[(size_t)r * F_OUT + c];
}

// ---------------- K3: out = relu( (M*e) @ agg / d ) ----------------
// Same pipelined tiling as K1; reads 8MB bitmask instead of 256MB A.
// e[] (32KB) is staged into LDS once via async copy (ASYNCcnt path) when available.
__global__ __launch_bounds__(256) void k3_attn(const uint32_t* __restrict__ mask,
                                               const float* __restrict__ ev,
                                               const __bf16* __restrict__ aggT,
                                               float* __restrict__ out) {
  __shared__ __bf16 lds_p[2][32 * 136];
  __shared__ float  lds_d[32];
  __shared__ float  lds_e[N_NODES];   // 32 KB: exp(sj - smax), whole vector
  const int t = threadIdx.x;
  const int w = t >> 5;
  const int l = t & 31;
  const int row0 = blockIdx.x * 32;

#if defined(USE_ASYNC_E)
  // async copy e[] (global -> LDS) : 256 threads x 8 x b128 = 32KB, tracked by ASYNCcnt
  #pragma unroll
  for (int i = 0; i < 8; ++i) {
    const int off = (t + 256 * i) * 4;  // float index, 16B aligned
    __builtin_amdgcn_global_load_async_to_lds_b128(
        (as1_i32x4_ptr)(uintptr_t)(ev + off),
        (as3_i32x4_ptr)(uint32_t)(uintptr_t)(lds_e + off),
        0, 0);
  }
#if __has_builtin(__builtin_amdgcn_s_wait_asynccnt)
  __builtin_amdgcn_s_wait_asynccnt(0);
#else
  asm volatile("s_wait_asynccnt 0x0" ::: "memory");
#endif
#else
  for (int i = t; i < N_NODES; i += 256) lds_e[i] = ev[i];
#endif
  __syncthreads();

  const int mtile = w >> 2;  // 0..1 (rows 0-15 / 16-31)
  const int ntile = w & 3;   // 0..3 (cols of agg, 16 each)
  f32x8 acc = {};
  float dacc[4] = {0.f, 0.f, 0.f, 0.f};

  const int arow = mtile * 16 + (l & 15);
  const int kb   = (l >> 4) * 8;
  const int bn   = ntile * 16 + (l & 15);
  const int bko  = (l >> 4) * 8;

  // prologue: mask words for tile 0
  uint4 mreg[2][4];
  #pragma unroll
  for (int it = 0; it < 4; ++it)
    mreg[0][it] = *(const uint4*)(mask + (size_t)(row0 + w + 8 * it) * 256);

  #pragma unroll 2
  for (int kk = 0; kk < N_NODES / 128; ++kk) {
    const int k = kk * 128;
    __bf16* buf = &lds_p[kk & 1][0];
    // ---- stage P tile kk: P[i,j] = bit ? e[j] : 0 ; accumulate denominator ----
    f32x4 e4 = *(const f32x4*)(lds_e + k + 4 * l);
    #pragma unroll
    for (int it = 0; it < 4; ++it) {
      const int rl = w + 8 * it;
      uint4 mw = mreg[kk & 1][it];
      float p0 = ((mw.x >> l) & 1u) ? e4.x : 0.f;
      float p1 = ((mw.y >> l) & 1u) ? e4.y : 0.f;
      float p2 = ((mw.z >> l) & 1u) ? e4.z : 0.f;
      float p3 = ((mw.w >> l) & 1u) ? e4.w : 0.f;
      dacc[it] += p0 + p1 + p2 + p3;
      bf16x4 pbv = {(__bf16)p0, (__bf16)p1, (__bf16)p2, (__bf16)p3};
      *(bf16x4*)(buf + rl * 136 + 4 * l) = pbv;
    }
    // ---- issue mask loads for tile kk+1 (stay in flight across the barrier) ----
    if (kk + 1 < N_NODES / 128) {
      #pragma unroll
      for (int it = 0; it < 4; ++it)
        mreg[(kk + 1) & 1][it] = *(const uint4*)(
            mask + (size_t)(row0 + w + 8 * it) * 256 + (kk + 1) * 4);
    }
    // ---- preload B operands (aggT) for tile kk ----
    bf16x16 bmat[4];
    #pragma unroll
    for (int s = 0; s < 4; ++s) {
      const __bf16* pb = aggT + (size_t)bn * N_NODES + k + 32 * s + bko;
      #pragma unroll
      for (int e2 = 0; e2 < 8; ++e2) { bmat[s][e2] = pb[e2]; bmat[s][e2 + 8] = pb[16 + e2]; }
    }
    wg_split_barrier();
    // ---- 4 WMMA K-substeps of 32 ----
    #pragma unroll
    for (int s = 0; s < 4; ++s) {
      const int ks = 32 * s;
      bf16x16 amat;
      const __bf16* pa = buf + arow * 136 + ks + kb;
      #pragma unroll
      for (int e2 = 0; e2 < 8; ++e2) { amat[e2] = pa[e2]; amat[e2 + 8] = pa[16 + e2]; }
      acc = __builtin_amdgcn_wmma_f32_16x16x32_bf16(false, amat, false, bmat[s],
                                                    (short)0, acc, false, false);
    }
  }

  // per-row softmax denominator: reduce dacc across the wave (one row per wave per it)
  #pragma unroll
  for (int it = 0; it < 4; ++it) {
    float v = dacc[it];
    for (int off = 16; off; off >>= 1) v += __shfl_xor(v, off, 32);
    if (l == 0) lds_d[w + 8 * it] = v;
  }
  __syncthreads();

  const int hb = (l >> 4) * 8;
  const int c = ntile * 16 + (l & 15);
  #pragma unroll
  for (int r = 0; r < 8; ++r) {
    int rl = mtile * 16 + r + hb;
    float h = fmaxf(acc[r] / lds_d[rl], 0.f);
    __builtin_nontemporal_store(h, out + (size_t)(row0 + rl) * F_OUT + c);
  }
}

extern "C" void kernel_launch(void* const* d_in, const int* in_sizes, int n_in,
                              void* d_out, int out_size, void* d_ws, size_t ws_size,
                              hipStream_t stream) {
  const float* A    = (const float*)d_in[0];
  const float* X    = (const float*)d_in[1];
  const float* W    = (const float*)d_in[2];
  const float* bias = (const float*)d_in[3];
  const float* phi  = (const float*)d_in[4];
  char* ws = (char*)d_ws;
  __bf16*   XT   = (__bf16*)(ws + WS_XT);
  uint32_t* mask = (uint32_t*)(ws + WS_MASK);
  float*    agg  = (float*)(ws + WS_AGG);
  __bf16*   aggT = (__bf16*)(ws + WS_AGGT);
  float*    sj   = (float*)(ws + WS_SJ);
  float*    e    = (float*)(ws + WS_E);
  float*    out  = (float*)d_out;

  hipLaunchKernelGGL(k0_xt,    dim3(N_NODES * F_IN / 256), dim3(256), 0, stream, X, XT);
  hipLaunchKernelGGL(k1_agg,   dim3(N_NODES / 32), dim3(256), 0, stream, A, XT, W, bias, mask, agg);
  hipLaunchKernelGGL(k2a_sj,   dim3(N_NODES / 256), dim3(256), 0, stream, agg, phi, sj);
  hipLaunchKernelGGL(k2b_exp,  dim3(1), dim3(256), 0, stream, sj, e);
  hipLaunchKernelGGL(k2c_aggT, dim3(N_NODES * F_OUT / 256), dim3(256), 0, stream, agg, aggT);
  hipLaunchKernelGGL(k3_attn,  dim3(N_NODES / 32), dim3(256), 0, stream, mask, e, aggT, out);
}
